// CRFOnLSTM_39917426049472
// MI455X (gfx1250) — matrix operand order repaired and checked by reading, not compiled, not run on hardware
//
#include <hip/hip_runtime.h>
#include <hip/hip_bf16.h>
#include <math.h>

// Problem constants (from reference)
#define B_    64
#define T_    256
#define LC_   16
#define CE_   100
#define WE_   300
#define CH_   100
#define WH_   300
#define NT_   20
#define START_ 18
#define STOP_  19
#define NEGV  (-10000.0f)
#define NBT_  (B_ * T_)   // 16384
#define MBLK  4           // 16x16 output tiles per wave along M

typedef __attribute__((ext_vector_type(16))) __bf16 v16bf;
typedef __attribute__((ext_vector_type(8)))  float  v8f;
typedef __attribute__((ext_vector_type(4)))  float  v4f;

// ---------------------------------------------------------------------------
// Fragment loaders for v_wmma_f32_16x16x32_bf16 (ISA 7.12.2 layouts).
// Full-tile loaders: unconditional 128-bit vector loads (global_load_b128).
// Tail loaders: branchless (address clamp + value select), no exec branching.
// ---------------------------------------------------------------------------
__device__ inline v16bf cvt16(v4f x0, v4f x1, v4f x2, v4f x3) {
  v16bf a;
#pragma unroll
  for (int e = 0; e < 4; ++e) a[e]      = (__bf16)x0[e];
#pragma unroll
  for (int e = 0; e < 4; ++e) a[4 + e]  = (__bf16)x1[e];
#pragma unroll
  for (int e = 0; e < 4; ++e) a[8 + e]  = (__bf16)x2[e];
#pragma unroll
  for (int e = 0; e < 4; ++e) a[12 + e] = (__bf16)x3[e];
  return a;
}

// A fragment 16x32 (MxK): elems 0-7 -> K=k0+half*8+e, elems 8-15 -> +16.
__device__ inline v16bf load_a_full(const float* __restrict__ row, int k0) {
  const int half = (threadIdx.x >> 4) & 1;
  const float* p0 = row + k0 + half * 8;
  const float* p1 = p0 + 16;
  return cvt16(*(const v4f*)p0, *(const v4f*)(p0 + 4),
               *(const v4f*)p1, *(const v4f*)(p1 + 4));
}

__device__ inline v16bf load_a_tail(const float* __restrict__ row, int k0, int K) {
  const int half = (threadIdx.x >> 4) & 1;
  v16bf a;
#pragma unroll
  for (int e = 0; e < 16; ++e) {
    int k = k0 + (e & 7) + ((e >> 3) << 4) + half * 8;
    int kc = (k < K) ? k : 0;           // clamp address (v_cndmask)
    float v = row[kc];                  // unconditional load
    a[e] = (__bf16)((k < K) ? v : 0.0f);
  }
  return a;
}

// Branchless loader for a full tile straddling a [rowA | rowB] concat boundary.
__device__ inline v16bf load_a_concat(const float* __restrict__ rowA, int lenA,
                                      const float* __restrict__ rowB, int k0) {
  const int half = (threadIdx.x >> 4) & 1;
  v16bf a;
#pragma unroll
  for (int e = 0; e < 16; ++e) {
    int k = k0 + (e & 7) + ((e >> 3) << 4) + half * 8;
    const float* p = (k < lenA) ? (rowA + k) : (rowB + (k - lenA));
    a[e] = (__bf16)(*p);                // unconditional load, selected address
  }
  return a;
}

// B fragment 32x16 (KxN), B[k][n] = W[n][k]: elems 0-7 -> K=k0+half*8+e, elems 8-15 -> +16.
__device__ inline v16bf load_b_full(const float* __restrict__ W, int ldw,
                                    int n0, int k0, int Nlim) {
  const int lane = threadIdx.x;
  int n = n0 + (lane & 15);
  n = (n < Nlim) ? n : (Nlim - 1);      // address clamp (cols >= Nlim unused)
  const int half = (lane >> 4) & 1;
  const float* p = W + (size_t)n * (size_t)ldw + k0 + half * 8;
  return cvt16(*(const v4f*)p, *(const v4f*)(p + 4),
               *(const v4f*)(p + 16), *(const v4f*)(p + 20));
}

__device__ inline v16bf load_b_tail(const float* __restrict__ W, int ldw,
                                    int n0, int k0, int K, int Nlim) {
  const int lane = threadIdx.x;
  int n = n0 + (lane & 15);
  n = (n < Nlim) ? n : (Nlim - 1);
  const int half = (lane >> 4) & 1;
  const float* row = W + (size_t)n * (size_t)ldw;
  v16bf b;
#pragma unroll
  for (int e = 0; e < 16; ++e) {
    int k = k0 + (e & 7) + ((e >> 3) << 4) + half * 8;
    int kc = (k < K) ? k : 0;
    float v = row[kc];
    b[e] = (__bf16)((k < K) ? v : 0.0f);
  }
  return b;
}

__device__ inline v8f wmma_acc(v16bf a, v16bf b, v8f c) {
  return __builtin_amdgcn_wmma_f32_16x16x32_bf16(false, a, false, b,
                                                 (short)0, c, false, false);
}

__device__ inline float sigmoidf_(float x) { return 1.0f / (1.0f + expf(-x)); }

// ---------------------------------------------------------------------------
// Char LSTM step: gates = char_emb[char_x[:,t]] @ cW_ih^T + h @ cW_hh^T + b
// One wave per 64x16 output strip (MBLK=4); gather fused into A loads.
// ---------------------------------------------------------------------------
__global__ void __launch_bounds__(32)
char_step_gemm(const int* __restrict__ char_x, const float* __restrict__ char_emb,
               const float* __restrict__ cW_ih, const float* __restrict__ cW_hh,
               const float* __restrict__ cb_ih, const float* __restrict__ cb_hh,
               const float* __restrict__ h, float* __restrict__ gates, int t) {
  const int lane = threadIdx.x;
  const int n0 = blockIdx.x * 16;
  const int m0 = blockIdx.y * (16 * MBLK);
  const int mr = lane & 15;
  const float* arow_x[MBLK];
  const float* arow_h[MBLK];
#pragma unroll
  for (int i = 0; i < MBLK; ++i) {
    int m = m0 + i * 16 + mr;
    int ci = char_x[(size_t)m * LC_ + t];
    arow_x[i] = char_emb + (size_t)ci * CE_;
    arow_h[i] = h + (size_t)m * CH_;
  }
  v8f acc[MBLK] = {};
  // x @ cW_ih^T : K=100 -> 3 full tiles + tail(4) at k0=96
  for (int k0 = 0; k0 + 32 <= CE_; k0 += 32) {
    v16bf b = load_b_full(cW_ih, CE_, n0, k0, 4 * CH_);
#pragma unroll
    for (int i = 0; i < MBLK; ++i)
      acc[i] = wmma_acc(load_a_full(arow_x[i], k0), b, acc[i]);
  }
  {
    const int k0 = (CE_ / 32) * 32;
    v16bf b = load_b_tail(cW_ih, CE_, n0, k0, CE_, 4 * CH_);
#pragma unroll
    for (int i = 0; i < MBLK; ++i)
      acc[i] = wmma_acc(load_a_tail(arow_x[i], k0, CE_), b, acc[i]);
  }
  // h @ cW_hh^T : K=100
  for (int k0 = 0; k0 + 32 <= CH_; k0 += 32) {
    v16bf b = load_b_full(cW_hh, CH_, n0, k0, 4 * CH_);
#pragma unroll
    for (int i = 0; i < MBLK; ++i)
      acc[i] = wmma_acc(load_a_full(arow_h[i], k0), b, acc[i]);
  }
  {
    const int k0 = (CH_ / 32) * 32;
    v16bf b = load_b_tail(cW_hh, CH_, n0, k0, CH_, 4 * CH_);
#pragma unroll
    for (int i = 0; i < MBLK; ++i)
      acc[i] = wmma_acc(load_a_tail(arow_h[i], k0, CH_), b, acc[i]);
  }
  const int n = n0 + mr;
  const float bias = cb_ih[n] + cb_hh[n];
  const int rbase = (lane >> 4) << 3;
#pragma unroll
  for (int i = 0; i < MBLK; ++i)
#pragma unroll
    for (int r = 0; r < 8; ++r)
      gates[(size_t)(m0 + i * 16 + rbase + r) * (4 * CH_) + n] = acc[i][r] + bias;
}

__global__ void char_cell(const int* __restrict__ char_x,
                          const float* __restrict__ gates,
                          float* __restrict__ h, float* __restrict__ c, int t) {
  int idx = blockIdx.x * blockDim.x + threadIdx.x;
  if (idx >= NBT_ * CH_) return;
  int m = idx / CH_, j = idx % CH_;
  const float* g = gates + (size_t)m * (4 * CH_);
  float i_ = sigmoidf_(g[j]);
  float f_ = sigmoidf_(g[CH_ + j]);
  float g_ = tanhf(g[2 * CH_ + j]);
  float o_ = sigmoidf_(g[3 * CH_ + j]);
  float cn = f_ * c[idx] + i_ * g_;
  float hn = o_ * tanhf(cn);
  bool mt = char_x[(size_t)m * LC_ + t] > 0;
  h[idx] = mt ? hn : h[idx];
  c[idx] = mt ? cn : c[idx];
}

// ---------------------------------------------------------------------------
// Word BiLSTM input projection: pre = [word_emb[word_x], c_h] @ W_ih^T + b
// M=16384, N=1200, K=400 (concat boundary at 300 handled per-tile).
// ---------------------------------------------------------------------------
__global__ void __launch_bounds__(32)
word_pre_gemm(const int* __restrict__ word_x, const float* __restrict__ word_emb,
              const float* __restrict__ ch, const float* __restrict__ W_ih,
              const float* __restrict__ b_ih, const float* __restrict__ b_hh,
              float* __restrict__ pre) {
  const int lane = threadIdx.x;
  const int n0 = blockIdx.x * 16;
  const int m0 = blockIdx.y * (16 * MBLK);
  const int mr = lane & 15;
  const float* erow[MBLK];
  const float* crow[MBLK];
#pragma unroll
  for (int i = 0; i < MBLK; ++i) {
    int m = m0 + i * 16 + mr;
    erow[i] = word_emb + (size_t)word_x[m] * WE_;
    crow[i] = ch + (size_t)m * CH_;
  }
  v8f acc[MBLK] = {};
  const int KTOT = WE_ + CH_;   // 400
  for (int k0 = 0; k0 + 32 <= KTOT; k0 += 32) {   // 12 full tiles
    v16bf b = load_b_full(W_ih, KTOT, n0, k0, 4 * WH_);
#pragma unroll
    for (int i = 0; i < MBLK; ++i) {
      v16bf a;
      if (k0 + 32 <= WE_)      a = load_a_full(erow[i], k0);
      else if (k0 >= WE_)      a = load_a_full(crow[i], k0 - WE_);
      else                     a = load_a_concat(erow[i], WE_, crow[i], k0);
      acc[i] = wmma_acc(a, b, acc[i]);
    }
  }
  {
    const int k0 = (KTOT / 32) * 32;   // 384 -> entirely in c_h region
    v16bf b = load_b_tail(W_ih, KTOT, n0, k0, KTOT, 4 * WH_);
#pragma unroll
    for (int i = 0; i < MBLK; ++i)
      acc[i] = wmma_acc(load_a_tail(crow[i], k0 - WE_, CH_), b, acc[i]);
  }
  const int n = n0 + mr;
  const float bias = b_ih[n] + b_hh[n];
  const int rbase = (lane >> 4) << 3;
#pragma unroll
  for (int i = 0; i < MBLK; ++i)
#pragma unroll
    for (int r = 0; r < 8; ++r)
      pre[(size_t)(m0 + i * 16 + rbase + r) * (4 * WH_) + n] = acc[i][r] + bias;
}

// Word recurrent step: gates = pre[:,t,:] + h @ W_hh^T. M=64, N=1200, K=300.
__global__ void __launch_bounds__(32)
word_step_gemm(const float* __restrict__ h, const float* __restrict__ W_hh,
               const float* __restrict__ pre, float* __restrict__ gates, int t) {
  const int lane = threadIdx.x;
  const int n0 = blockIdx.x * 16;
  const int m0 = blockIdx.y * (16 * MBLK);
  const int mr = lane & 15;
  const float* arow[MBLK];
#pragma unroll
  for (int i = 0; i < MBLK; ++i) arow[i] = h + (size_t)(m0 + i * 16 + mr) * WH_;
  v8f acc[MBLK] = {};
  for (int k0 = 0; k0 + 32 <= WH_; k0 += 32) {    // 9 full tiles
    v16bf b = load_b_full(W_hh, WH_, n0, k0, 4 * WH_);
#pragma unroll
    for (int i = 0; i < MBLK; ++i)
      acc[i] = wmma_acc(load_a_full(arow[i], k0), b, acc[i]);
  }
  {
    const int k0 = (WH_ / 32) * 32;   // 288, len 12
    v16bf b = load_b_tail(W_hh, WH_, n0, k0, WH_, 4 * WH_);
#pragma unroll
    for (int i = 0; i < MBLK; ++i)
      acc[i] = wmma_acc(load_a_tail(arow[i], k0, WH_), b, acc[i]);
  }
  const int n = n0 + mr;
  const int rbase = (lane >> 4) << 3;
#pragma unroll
  for (int i = 0; i < MBLK; ++i)
#pragma unroll
    for (int r = 0; r < 8; ++r) {
      int mm = m0 + i * 16 + rbase + r;
      gates[(size_t)mm * (4 * WH_) + n] =
          acc[i][r] + pre[((size_t)mm * T_ + t) * (4 * WH_) + n];
    }
}

__global__ void word_cell(const int* __restrict__ word_x,
                          const float* __restrict__ gates,
                          float* __restrict__ h, float* __restrict__ c,
                          float* __restrict__ out, int t) {
  int idx = blockIdx.x * blockDim.x + threadIdx.x;
  if (idx >= B_ * WH_) return;
  int m = idx / WH_, j = idx % WH_;
  const float* g = gates + (size_t)m * (4 * WH_);
  float i_ = sigmoidf_(g[j]);
  float f_ = sigmoidf_(g[WH_ + j]);
  float g_ = tanhf(g[2 * WH_ + j]);
  float o_ = sigmoidf_(g[3 * WH_ + j]);
  float cn = f_ * c[idx] + i_ * g_;
  float hn = o_ * tanhf(cn);
  bool mt = word_x[(size_t)m * T_ + t] > 0;
  out[((size_t)m * T_ + t) * WH_ + j] = mt ? hn : 0.0f;
  h[idx] = mt ? hn : h[idx];
  c[idx] = mt ? cn : c[idx];
}

// ---------------------------------------------------------------------------
// Output projection: logits = [f_out, b_out] @ out_W^T + out_b, masked.
// M=16384, N=20 (two 16-wide tiles, address-clamped), K=600.
// ---------------------------------------------------------------------------
__global__ void __launch_bounds__(32)
out_proj_gemm(const int* __restrict__ word_x, const float* __restrict__ fo,
              const float* __restrict__ bo, const float* __restrict__ out_W,
              const float* __restrict__ out_b, float* __restrict__ logits) {
  const int lane = threadIdx.x;
  const int n0 = blockIdx.x * 16;
  const int m0 = blockIdx.y * (16 * MBLK);
  const int mr = lane & 15;
  const float* frow[MBLK];
  const float* brow[MBLK];
#pragma unroll
  for (int i = 0; i < MBLK; ++i) {
    int m = m0 + i * 16 + mr;
    frow[i] = fo + (size_t)m * WH_;
    brow[i] = bo + (size_t)m * WH_;
  }
  v8f acc[MBLK] = {};
  const int KTOT = 2 * WH_;   // 600
  for (int k0 = 0; k0 + 32 <= KTOT; k0 += 32) {   // 18 full tiles
    v16bf b = load_b_full(out_W, KTOT, n0, k0, NT_);
#pragma unroll
    for (int i = 0; i < MBLK; ++i) {
      v16bf a;
      if (k0 + 32 <= WH_)      a = load_a_full(frow[i], k0);
      else if (k0 >= WH_)      a = load_a_full(brow[i], k0 - WH_);
      else                     a = load_a_concat(frow[i], WH_, brow[i], k0);
      acc[i] = wmma_acc(a, b, acc[i]);
    }
  }
  {
    const int k0 = (KTOT / 32) * 32;   // 576 -> entirely in b_out region
    v16bf b = load_b_tail(out_W, KTOT, n0, k0, KTOT, NT_);
#pragma unroll
    for (int i = 0; i < MBLK; ++i)
      acc[i] = wmma_acc(load_a_tail(brow[i], k0 - WH_, WH_), b, acc[i]);
  }
  const int n = n0 + mr;
  if (n < NT_) {
    const float bias = out_b[n];
    const int rbase = (lane >> 4) << 3;
#pragma unroll
    for (int i = 0; i < MBLK; ++i)
#pragma unroll
      for (int r = 0; r < 8; ++r) {
        int mm = m0 + i * 16 + rbase + r;
        bool mt = word_x[mm] > 0;
        logits[(size_t)mm * NT_ + n] = mt ? (acc[i][r] + bias) : 0.0f;
      }
  }
}

// ---------------------------------------------------------------------------
// CRF forward scan: single workgroup, alpha + transitions resident in LDS.
// ---------------------------------------------------------------------------
__global__ void __launch_bounds__(640)
crf_alpha(const int* __restrict__ word_x, const float* __restrict__ logits,
          const float* __restrict__ tr, float* __restrict__ Z) {
  __shared__ float alpha[B_][NT_];
  __shared__ float trs[NT_][NT_];
  const int tid = threadIdx.x;
  if (tid < NT_ * NT_) trs[tid / NT_][tid % NT_] = tr[tid];
  for (int p = tid; p < B_ * NT_; p += 640)
    alpha[p / NT_][p % NT_] = ((p % NT_) == START_) ? 0.0f : NEGV;
  __syncthreads();
  for (int t = 0; t < T_; ++t) {
    float nv[2];
#pragma unroll
    for (int q = 0; q < 2; ++q) {
      int p = tid + q * 640;
      int b = p / NT_, j = p % NT_;
      float mx = -3.4e38f;
      for (int k = 0; k < NT_; ++k) mx = fmaxf(mx, alpha[b][k] + trs[j][k]);
      float s = 0.0f;
      for (int k = 0; k < NT_; ++k) s += expf(alpha[b][k] + trs[j][k] - mx);
      float a_t = logits[((size_t)b * T_ + t) * NT_ + j] + mx + logf(s);
      bool mt = word_x[(size_t)b * T_ + t] > 0;
      nv[q] = mt ? a_t : alpha[b][j];
    }
    __syncthreads();
#pragma unroll
    for (int q = 0; q < 2; ++q) {
      int p = tid + q * 640;
      alpha[p / NT_][p % NT_] = nv[q];
    }
    __syncthreads();
  }
  if (tid < B_) {
    float mx = -3.4e38f;
    for (int j = 0; j < NT_; ++j) mx = fmaxf(mx, alpha[tid][j] + trs[STOP_][j]);
    float s = 0.0f;
    for (int j = 0; j < NT_; ++j) s += expf(alpha[tid][j] + trs[STOP_][j] - mx);
    Z[tid] = mx + logf(s);
  }
}

// Gold-path score + final output: out[b] = Z[b] - score[b].
__global__ void __launch_bounds__(256)
score_out(const int* __restrict__ word_x, const int* __restrict__ y,
          const float* __restrict__ logits, const float* __restrict__ tr,
          const float* __restrict__ Z, float* __restrict__ out) {
  const int b = blockIdx.x;
  const int tid = threadIdx.x;
  __shared__ float ssum[256];
  __shared__ int   slen[256];
  float partial = 0.0f;
  int lenp = 0;
  for (int t = tid; t < T_; t += 256) {
    bool mt = word_x[(size_t)b * T_ + t] > 0;
    if (mt) {
      int yt = y[(size_t)b * T_ + t];
      int prev = (t == 0) ? START_ : y[(size_t)b * T_ + t - 1];
      partial += logits[((size_t)b * T_ + t) * NT_ + yt] + tr[yt * NT_ + prev];
      lenp += 1;
    }
  }
  ssum[tid] = partial; slen[tid] = lenp;
  __syncthreads();
  for (int s = 128; s > 0; s >>= 1) {
    if (tid < s) { ssum[tid] += ssum[tid + s]; slen[tid] += slen[tid + s]; }
    __syncthreads();
  }
  if (tid == 0) {
    int len = slen[0];
    int last = (len > 0) ? y[(size_t)b * T_ + len - 1] : START_;
    out[b] = Z[b] - (ssum[0] + tr[STOP_ * NT_ + last]);
  }
}

// ---------------------------------------------------------------------------
// Host-side orchestration (graph-capturable: launches + memsetAsync only)
// ---------------------------------------------------------------------------
extern "C" void kernel_launch(void* const* d_in, const int* in_sizes, int n_in,
                              void* d_out, int out_size, void* d_ws, size_t ws_size,
                              hipStream_t stream) {
  const int*   word_x   = (const int*)  d_in[0];
  const int*   char_x   = (const int*)  d_in[1];
  const int*   y        = (const int*)  d_in[2];
  const float* word_emb = (const float*)d_in[3];
  const float* char_emb = (const float*)d_in[4];
  const float* cW_ih    = (const float*)d_in[5];
  const float* cW_hh    = (const float*)d_in[6];
  const float* cb_ih    = (const float*)d_in[7];
  const float* cb_hh    = (const float*)d_in[8];
  const float* fW_ih    = (const float*)d_in[9];
  const float* fW_hh    = (const float*)d_in[10];
  const float* fb_ih    = (const float*)d_in[11];
  const float* fb_hh    = (const float*)d_in[12];
  const float* bW_ih    = (const float*)d_in[13];
  const float* bW_hh    = (const float*)d_in[14];
  const float* bb_ih    = (const float*)d_in[15];
  const float* bb_hh    = (const float*)d_in[16];
  const float* out_W    = (const float*)d_in[17];
  const float* out_b    = (const float*)d_in[18];
  const float* trans    = (const float*)d_in[19];

  // Workspace layout (floats), all offsets 16B-aligned. ~238 MB total.
  float* ws = (float*)d_ws;
  size_t off = 0;
  float* h_c    = ws + off; off += (size_t)NBT_ * CH_;
  float* c_c    = ws + off; off += (size_t)NBT_ * CH_;
  float* h_f    = ws + off; off += (size_t)B_ * WH_;
  float* c_f    = ws + off; off += (size_t)B_ * WH_;
  float* h_b    = ws + off; off += (size_t)B_ * WH_;
  float* c_b    = ws + off; off += (size_t)B_ * WH_;
  float* gates_c= ws + off; off += (size_t)NBT_ * 4 * CH_;
  float* pre_f  = ws + off; off += (size_t)NBT_ * 4 * WH_;
  float* pre_b  = ws + off; off += (size_t)NBT_ * 4 * WH_;
  float* f_out  = ws + off; off += (size_t)NBT_ * WH_;
  float* b_out  = ws + off; off += (size_t)NBT_ * WH_;
  float* logits = ws + off; off += (size_t)NBT_ * NT_;
  float* gates_f= ws + off; off += (size_t)B_ * 4 * WH_;
  float* gates_b= ws + off; off += (size_t)B_ * 4 * WH_;
  float* Zv     = ws + off; off += (size_t)B_;

  // Zero-init carried LSTM state (ws is poisoned by the harness).
  hipMemsetAsync(h_c, 0, (size_t)2 * NBT_ * CH_ * sizeof(float), stream);   // h_c, c_c
  hipMemsetAsync(h_f, 0, (size_t)4 * B_ * WH_ * sizeof(float), stream);     // h_f,c_f,h_b,c_b

  // --- Char LSTM (16 sequential steps, fused x+h projection GEMM) ---
  for (int t = 0; t < LC_; ++t) {
    char_step_gemm<<<dim3(4 * CH_ / 16, NBT_ / (16 * MBLK)), 32, 0, stream>>>(
        char_x, char_emb, cW_ih, cW_hh, cb_ih, cb_hh, h_c, gates_c, t);
    char_cell<<<(NBT_ * CH_ + 255) / 256, 256, 0, stream>>>(
        char_x, gates_c, h_c, c_c, t);
  }

  // --- Word BiLSTM input projections (big parallel GEMMs) ---
  word_pre_gemm<<<dim3(4 * WH_ / 16, NBT_ / (16 * MBLK)), 32, 0, stream>>>(
      word_x, word_emb, h_c, fW_ih, fb_ih, fb_hh, pre_f);
  word_pre_gemm<<<dim3(4 * WH_ / 16, NBT_ / (16 * MBLK)), 32, 0, stream>>>(
      word_x, word_emb, h_c, bW_ih, bb_ih, bb_hh, pre_b);

  // --- Word BiLSTM recurrence (256 sequential steps each direction) ---
  for (int t = 0; t < T_; ++t) {
    word_step_gemm<<<dim3(4 * WH_ / 16, B_ / (16 * MBLK)), 32, 0, stream>>>(
        h_f, fW_hh, pre_f, gates_f, t);
    word_cell<<<(B_ * WH_ + 255) / 256, 256, 0, stream>>>(
        word_x, gates_f, h_f, c_f, f_out, t);
    int tb = T_ - 1 - t;
    word_step_gemm<<<dim3(4 * WH_ / 16, B_ / (16 * MBLK)), 32, 0, stream>>>(
        h_b, bW_hh, pre_b, gates_b, tb);
    word_cell<<<(B_ * WH_ + 255) / 256, 256, 0, stream>>>(
        word_x, gates_b, h_b, c_b, b_out, tb);
  }

  // --- Tag logits (N=20 -> two 16-wide WMMA tiles, address-clamped) ---
  out_proj_gemm<<<dim3(2, NBT_ / (16 * MBLK)), 32, 0, stream>>>(
      word_x, f_out, b_out, out_W, out_b, logits);

  // --- CRF forward scan + partition function ---
  crf_alpha<<<1, 640, 0, stream>>>(word_x, logits, trans, Zv);

  // --- Gold score and final output ---
  score_out<<<B_, 256, 0, stream>>>(word_x, y, logits, trans, Zv, (float*)d_out);
}
